// CrossAttention_6133213298770
// MI455X (gfx1250) — matrix operand order repaired and checked by reading, not compiled, not run on hardware
//
#include <hip/hip_runtime.h>
#include <hip/hip_bf16.h>

// ---------------- problem constants (match reference) ----------------
constexpr int Bc = 4;
constexpr int Sc = 2048;
constexpr int Hc = 1024;
constexpr int NHc = 16;
constexpr int HDc = 64;         // head dim
constexpr int Mrows = Bc * Sc;  // 8192

typedef __bf16 bf16_t;
typedef __attribute__((ext_vector_type(16))) __bf16 v16bf;
typedef __attribute__((ext_vector_type(8)))  float  v8f;
typedef __attribute__((ext_vector_type(4)))  int    v4i;

#define GAS __attribute__((address_space(1)))
#define LAS __attribute__((address_space(3)))

#if __has_builtin(__builtin_amdgcn_global_load_async_to_lds_b128)
#define HAVE_ASYNC_LDS 1
#else
#define HAVE_ASYNC_LDS 0
#endif

__device__ __forceinline__ void wait_asynccnt0() {
#if __has_builtin(__builtin_amdgcn_s_wait_asynccnt)
    __builtin_amdgcn_s_wait_asynccnt(0);
#else
    asm volatile("s_wait_asynccnt 0" ::: "memory");
#endif
}

// A-matrix 16-bit 16x32 per-lane K index map (contiguous, matches B fill):
// lane<16 holds K {0..15}; lane>=16 holds K {16..31}, 2 per VGPR.
__device__ __forceinline__ int a_kmap(int i, int h) {
    return 16 * h + 2 * i;
}

// Load A fragment (16x32 bf16) from row-major LDS tile (rows = lane%16).
__device__ __forceinline__ v16bf load_a_frag(const bf16_t* tile, int lda, int lane) {
    int r = lane & 15, h = lane >> 4;
    const bf16_t* row = tile + r * lda;
    v16bf a;
#pragma unroll
    for (int i = 0; i < 8; ++i) {
        int k = a_kmap(i, h);
        a[2 * i]     = row[k];
        a[2 * i + 1] = row[k + 1];
    }
    return a;
}

// Load B fragment (32x16 bf16) from LDS tile stored [K][N] row-major.
// lane<16: N=lane, K=0..15 ; lane>=16: N=lane-16, K=16..31 (2 per VGPR).
__device__ __forceinline__ v16bf load_b_frag(const bf16_t* tile, int ldb, int lane) {
    int n = lane & 15, h = lane >> 4;
    v16bf b;
#pragma unroll
    for (int i = 0; i < 8; ++i) {
        int k = 16 * h + 2 * i;
        b[2 * i]     = tile[k * ldb + n];
        b[2 * i + 1] = tile[(k + 1) * ldb + n];
    }
    return b;
}

// ---------------- projection GEMM: out = A[MxK] * W[KxN] + bias -------------
// 128x128 block tile, 8 waves, each wave owns a 32x64 sub-tile (8 WMMAs/k-step).
template <typename AT, bool OUT_F32>
__global__ __launch_bounds__(256) void gemm_wmma(const AT* __restrict__ A,
                                                 const float* __restrict__ W,
                                                 const float* __restrict__ bias,
                                                 void* __restrict__ outp,
                                                 int M, int N, int K) {
    constexpr int BM = 128, BN = 128, BK = 32;
    __shared__ bf16_t sA[BM][BK + 8];
    __shared__ bf16_t sB[BK][BN + 8];

    const int tid = threadIdx.x, lane = tid & 31, wave = tid >> 5;
    const int wm = wave & 3;   // 4 row-waves  -> 32 rows each
    const int wn = wave >> 2;  // 2 col-waves  -> 64 cols each
    const int m0 = blockIdx.y * BM, n0 = blockIdx.x * BN;

    v8f acc[2][4] = {};

    for (int k0 = 0; k0 < K; k0 += BK) {
        // stage A (128x32) and B (32x128) as bf16; 4-wide chunks -> b128/b64 loads
#pragma unroll
        for (int e = tid * 4; e < BM * BK; e += 256 * 4) {
            int r = e >> 5, c = e & 31;
#pragma unroll
            for (int u = 0; u < 4; ++u)
                sA[r][c + u] = (bf16_t)(float)A[(m0 + r) * K + (k0 + c + u)];
        }
#pragma unroll
        for (int e = tid * 4; e < BK * BN; e += 256 * 4) {
            int r = e >> 7, c = e & 127;
#pragma unroll
            for (int u = 0; u < 4; ++u)
                sB[r][c + u] = (bf16_t)W[(k0 + r) * N + (n0 + c + u)];
        }
        if (k0 + BK < K) {  // speculative prefetch of next tiles -> global_prefetch_b8
            __builtin_prefetch(&A[(m0 + (tid >> 1)) * K + (k0 + BK)], 0, 0);
            __builtin_prefetch(&W[(k0 + BK + (tid >> 4)) * N + n0 + ((tid & 15) << 3)], 0, 0);
        }
        __syncthreads();

        v16bf af[2], bf[4];
#pragma unroll
        for (int i = 0; i < 2; ++i)
            af[i] = load_a_frag(&sA[32 * wm + 16 * i][0], BK + 8, lane);
#pragma unroll
        for (int j = 0; j < 4; ++j)
            bf[j] = load_b_frag(&sB[0][64 * wn + 16 * j], BN + 8, lane);
#pragma unroll
        for (int i = 0; i < 2; ++i)
#pragma unroll
            for (int j = 0; j < 4; ++j)
                acc[i][j] = __builtin_amdgcn_wmma_f32_16x16x32_bf16(
                    false, af[i], false, bf[j], (short)0, acc[i][j], false, false);
        __syncthreads();
    }

    // epilogue: C layout -> row = 8*h + jr, col = lane%16
    const int h = lane >> 4, nn = lane & 15;
#pragma unroll
    for (int i = 0; i < 2; ++i) {
#pragma unroll
        for (int j = 0; j < 4; ++j) {
            int cb = n0 + 64 * wn + 16 * j;
            float bv = bias[cb + nn];
#pragma unroll
            for (int jr = 0; jr < 8; ++jr) {
                int row = m0 + 32 * wm + 16 * i + 8 * h + jr;
                float v = acc[i][j][jr] + bv;
                if constexpr (OUT_F32)
                    ((float*)outp)[row * N + cb + nn] = v;
                else
                    ((bf16_t*)outp)[row * N + cb + nn] = (bf16_t)v;
            }
        }
    }
}

// ---------------- flash attention (bf16 QK^T / PV, f32 softmax) ----------------
// Q: [B,S,H] bf16 ; KV: [B,S,2H] bf16 (k at h*64+d, v at H+h*64+d) ; O: [B,S,H] bf16
__global__ __launch_bounds__(256) void flash_attn(const bf16_t* __restrict__ Q,
                                                  const bf16_t* __restrict__ KV,
                                                  const unsigned char* __restrict__ kpad,
                                                  bf16_t* __restrict__ O) {
    constexpr int BQ = 128, BK = 32;
    const int b = blockIdx.z, head = blockIdx.y, q0 = blockIdx.x * BQ;
    const int tid = threadIdx.x, lane = tid & 31, wave = tid >> 5;
    const int qw = q0 + 16 * wave;  // this wave's 16 query rows
    const int r = lane & 15, h = lane >> 4;

    __shared__ bf16_t sK[BK][HDc + 8];
    __shared__ bf16_t sV[BK][HDc + 8];
    __shared__ bf16_t sP[8][16][BK + 8];  // per-wave P bounce (C-layout -> A-layout)

    // Q fragments for HD chunks [0,32) and [32,64): loaded once, stay in VGPRs.
    const bf16_t* qrow = Q + ((b * Sc + qw + r) * Hc + head * HDc);
    v16bf qf[2];
#pragma unroll
    for (int c = 0; c < 2; ++c) {
#pragma unroll
        for (int i = 0; i < 8; ++i) {
            int k = c * 32 + a_kmap(i, h);
            qf[c][2 * i]     = qrow[k];
            qf[c][2 * i + 1] = qrow[k + 1];
        }
    }

    v8f acc[4] = {v8f{}, v8f{}, v8f{}, v8f{}};  // [hd 16-chunk][row]
    float mrow[8], lrow[8];
#pragma unroll
    for (int j = 0; j < 8; ++j) { mrow[j] = -3.0e38f; lrow[j] = 0.0f; }

    const float scale = 0.125f;          // 1/sqrt(64)
    const int nChunks = (q0 + BQ) / BK;  // causal bound for the whole block

    // this thread's slice of the K/V staging: one b128 (8 bf16) per tile
    const int skr = tid >> 3, sch = (tid & 7) * 8;

    for (int jc = 0; jc < nChunks; ++jc) {
        const int kbase = jc * BK;
        const bf16_t* kvrow = KV + ((b * Sc + kbase + skr) * 2 * Hc + head * HDc + sch);
#if HAVE_ASYNC_LDS
        // CDNA5 async copy global -> LDS (ASYNCcnt), no VGPR round-trip
        __builtin_amdgcn_global_load_async_to_lds_b128(
            (GAS v4i*)(kvrow), (LAS v4i*)&sK[skr][sch], 0, 0);
        __builtin_amdgcn_global_load_async_to_lds_b128(
            (GAS v4i*)(kvrow + Hc), (LAS v4i*)&sV[skr][sch], 0, 0);
        wait_asynccnt0();
#else
#pragma unroll
        for (int u = 0; u < 8; ++u) {
            sK[skr][sch + u] = kvrow[u];
            sV[skr][sch + u] = kvrow[Hc + u];
        }
#endif
        __syncthreads();

        // scores S = Q * K^T  (two 16-key halves, K-depth = HD = 2 chunks of 32)
        v8f s0 = {}, s1 = {};
#pragma unroll
        for (int c = 0; c < 2; ++c) {
            v16bf bk0, bk1;
#pragma unroll
            for (int i = 0; i < 8; ++i) {
                int kk = c * 32 + 16 * h + 2 * i;
                bk0[2 * i]     = sK[r][kk];       // key n = r       (cols 0..15)
                bk0[2 * i + 1] = sK[r][kk + 1];
                bk1[2 * i]     = sK[16 + r][kk];  // key n = 16 + r  (cols 16..31)
                bk1[2 * i + 1] = sK[16 + r][kk + 1];
            }
            s0 = __builtin_amdgcn_wmma_f32_16x16x32_bf16(false, qf[c], false, bk0, (short)0, s0, false, false);
            s1 = __builtin_amdgcn_wmma_f32_16x16x32_bf16(false, qf[c], false, bk1, (short)0, s1, false, false);
        }

        // masking + online softmax; s-reg j holds row 8h+j across the 16-lane half
        const int k0i = kbase + r, k1i = kbase + 16 + r;
        const bool pad0 = kpad[b * Sc + k0i] != 0;
        const bool pad1 = kpad[b * Sc + k1i] != 0;
#pragma unroll
        for (int j = 0; j < 8; ++j) {
            int qidx = qw + 8 * h + j;
            float v0 = (k0i > qidx || pad0) ? -3.0e38f : s0[j] * scale;
            float v1 = (k1i > qidx || pad1) ? -3.0e38f : s1[j] * scale;
            float rmax = fmaxf(v0, v1);
#pragma unroll
            for (int d = 1; d < 16; d <<= 1)
                rmax = fmaxf(rmax, __shfl_xor(rmax, d, 32));  // stays within 16-lane half
            float mnew = fmaxf(mrow[j], rmax);
            float corr = __expf(mrow[j] - mnew);
            float p0 = __expf(v0 - mnew);
            float p1 = __expf(v1 - mnew);
            float rsum = p0 + p1;
#pragma unroll
            for (int d = 1; d < 16; d <<= 1)
                rsum += __shfl_xor(rsum, d, 32);
            lrow[j] = lrow[j] * corr + rsum;
            mrow[j] = mnew;
#pragma unroll
            for (int t = 0; t < 4; ++t) acc[t][j] *= corr;
            sP[wave][8 * h + j][r]      = (bf16_t)p0;
            sP[wave][8 * h + j][16 + r] = (bf16_t)p1;
        }
        __syncthreads();  // P visible across lanes (and sK reads complete)

        // O += P * V : A = P (16x32), B = V (32 keys x 64 hd)
        v16bf pf;
#pragma unroll
        for (int i = 0; i < 8; ++i) {
            int k = a_kmap(i, h);
            pf[2 * i]     = sP[wave][r][k];
            pf[2 * i + 1] = sP[wave][r][k + 1];
        }
#pragma unroll
        for (int t = 0; t < 4; ++t) {
            v16bf vf;
            int nhd = t * 16 + r;
#pragma unroll
            for (int i = 0; i < 8; ++i) {
                int kk = 16 * h + 2 * i;
                vf[2 * i]     = sV[kk][nhd];
                vf[2 * i + 1] = sV[kk + 1][nhd];
            }
            acc[t] = __builtin_amdgcn_wmma_f32_16x16x32_bf16(false, pf, false, vf, (short)0, acc[t], false, false);
        }
        __syncthreads();  // sV reads complete before next-iteration overwrite
    }

    // epilogue: normalize and store bf16 attention output in [B,S,H] layout
#pragma unroll
    for (int j = 0; j < 8; ++j) {
        int row = qw + 8 * h + j;
        float inv_l = 1.0f / lrow[j];
#pragma unroll
        for (int t = 0; t < 4; ++t) {
            float val = acc[t][j] * inv_l;
            O[(b * Sc + row) * Hc + head * HDc + t * 16 + r] = (bf16_t)val;
        }
    }
}

// ---------------- host-side launch ----------------
extern "C" void kernel_launch(void* const* d_in, const int* in_sizes, int n_in,
                              void* d_out, int out_size, void* d_ws, size_t ws_size,
                              hipStream_t stream) {
    (void)in_sizes; (void)n_in; (void)out_size; (void)ws_size;
    const float* x     = (const float*)d_in[0];
    const float* y     = (const float*)d_in[1];
    const unsigned char* mask = (const unsigned char*)d_in[2];  // jax bool = 1 byte
    const float* Wq_w  = (const float*)d_in[3];
    const float* Wq_b  = (const float*)d_in[4];
    const float* Wkv_w = (const float*)d_in[5];
    const float* Wkv_b = (const float*)d_in[6];
    const float* Wo_w  = (const float*)d_in[7];
    const float* Wo_b  = (const float*)d_in[8];

    bf16_t* qb   = (bf16_t*)d_ws;                 // [B,S,H]
    bf16_t* kvb  = qb + (size_t)Mrows * Hc;       // [B,S,2H]
    bf16_t* attn = kvb + (size_t)Mrows * 2 * Hc;  // [B,S,H]

    dim3 blk(256);
    // Q = x @ Wq + b  -> bf16
    gemm_wmma<float, false><<<dim3(Hc / 128, Mrows / 128), blk, 0, stream>>>(
        x, Wq_w, Wq_b, (void*)qb, Mrows, Hc, Hc);
    // KV = y @ Wkv + b -> bf16
    gemm_wmma<float, false><<<dim3(2 * Hc / 128, Mrows / 128), blk, 0, stream>>>(
        y, Wkv_w, Wkv_b, (void*)kvb, Mrows, 2 * Hc, Hc);
    // flash attention
    flash_attn<<<dim3(Sc / 128, NHc, Bc), blk, 0, stream>>>(qb, kvb, mask, attn);
    // out = attn @ Wo + b -> f32
    gemm_wmma<bf16_t, true><<<dim3(Hc / 128, Mrows / 128), blk, 0, stream>>>(
        attn, Wo_w, Wo_b, d_out, Mrows, Hc, Hc);
}
